// FullOnlineESPPLayer_11708080849226
// MI455X (gfx1250) — compile-verified, matrix operand order checked by reading, not run on hardware
//
#include <hip/hip_runtime.h>
#include <hip/hip_bf16.h>

typedef __bf16 bf16_t;
typedef __attribute__((ext_vector_type(16))) __bf16 v16bf;
typedef __attribute__((ext_vector_type(8)))  float  v8f;
typedef __attribute__((ext_vector_type(4)))  unsigned int v4u;
typedef __attribute__((ext_vector_type(8)))  int v8i;
typedef __attribute__((ext_vector_type(4)))  int v4i;

#define BSZ 256
#define LDIM 4096
#define NDIM 4096
#define BETA 0.95f
#define GAMMA 0.1f
#define LR 0.01f
#define INPUT_THR 0.02f
#define PI_F 3.14159265358979f

// LDS tile geometry for K1: 64 rows x 32 f32, TDM-padded to 36-float row
// stride (pad 4 DWORDs every 32 DWORDs) -> 144B rows: 16B-aligned b128 LDS
// loads, and half-wave row accesses land on 16 distinct banks (36 mod 64).
#define TROWS 64
#define TK 32
#define TSTRIDE 36
#define TILE_FLTS (TROWS * TSTRIDE)   // 2304 floats per buffer

// ---------------------------------------------------------------------------
// WMMA helpers (16-bit A/B layout, CDNA5 ISA 7.12.2): lane ln holds row
// (ln&15) K-chunks [8h,8h+8) and [16+8h,16+8h+8), h = ln>>4.
// ---------------------------------------------------------------------------
__device__ __forceinline__ v16bf load_row_f32_as_bf16(const float* rowbase, int h) {
  const float* p = rowbase + 8 * h;
  const float* q = rowbase + 16 + 8 * h;
  float4 a0 = *(const float4*)(p);
  float4 a1 = *(const float4*)(p + 4);
  float4 b0 = *(const float4*)(q);
  float4 b1 = *(const float4*)(q + 4);
  v16bf r;
  r[0]  = (__bf16)a0.x; r[1]  = (__bf16)a0.y; r[2]  = (__bf16)a0.z; r[3]  = (__bf16)a0.w;
  r[4]  = (__bf16)a1.x; r[5]  = (__bf16)a1.y; r[6]  = (__bf16)a1.z; r[7]  = (__bf16)a1.w;
  r[8]  = (__bf16)b0.x; r[9]  = (__bf16)b0.y; r[10] = (__bf16)b0.z; r[11] = (__bf16)b0.w;
  r[12] = (__bf16)b1.x; r[13] = (__bf16)b1.y; r[14] = (__bf16)b1.z; r[15] = (__bf16)b1.w;
  return r;
}

__device__ __forceinline__ v16bf load_row_bf16(const bf16_t* rowbase, int h) {
  union { v16bf v; uint4 q[2]; } u;
  u.q[0] = *(const uint4*)(rowbase + 8 * h);
  u.q[1] = *(const uint4*)(rowbase + 16 + 8 * h);
  return u.v;
}

__device__ __forceinline__ v8f wmma_bf16(v16bf a, v16bf b, v8f c) {
  return __builtin_amdgcn_wmma_f32_16x16x32_bf16(
      false, a, false, b, (short)0, c, false, false);
}

// ---------------------------------------------------------------------------
// TDM 2D tile load: build D# (ISA 8.3/8.4) and issue tensor_load_to_lds.
// group0: count=1 | lds_addr | 57-bit global addr | type=2
// group1: data_size=4B, pad_enable, pad_interval=32dw, pad_amount=4dw,
//         tensor dims/strides, tile dims.  groups 2/3 (+spare) = 0 (2D).
// amdgpu-toolchain builtin arity: (v4u, v8i, v4i, v4i, v8i, i32 cpol).
// ---------------------------------------------------------------------------
__device__ __forceinline__ void tdm_load_2d(const float* gbase,
                                            unsigned lds_byte_addr,
                                            unsigned rows, unsigned kelems,
                                            unsigned row_stride_elems) {
  unsigned long long ga = (unsigned long long)(const void*)gbase;
  v4u g0;
  g0[0] = 1u;                                       // count=1 (valid user D#)
  g0[1] = lds_byte_addr;                            // LDS destination
  g0[2] = (unsigned)(ga & 0xFFFFFFFFu);             // global_addr[31:0]
  g0[3] = (unsigned)((ga >> 32) & 0x01FFFFFFu) | (2u << 30);  // addr[56:32]|type=2
  v8i g1;
  g1[0] = (int)0x07120000u;                         // dsz=4B,pad_en,int=32dw,amt=4dw
  g1[1] = (int)((row_stride_elems & 0xFFFFu) << 16);        // tensor_dim0[15:0]
  g1[2] = (int)(((row_stride_elems >> 16) & 0xFFFFu) |
                ((rows & 0xFFFFu) << 16));                  // dim0 hi | dim1 lo
  g1[3] = (int)(((rows >> 16) & 0xFFFFu) | (kelems << 16)); // dim1 hi | tile_dim0
  g1[4] = (int)(rows & 0xFFFFu);                            // tile_dim1 (tile_dim2=0)
  g1[5] = (int)row_stride_elems;                            // tensor_dim0_stride lo32
  g1[6] = 0;
  g1[7] = 0;
  v4i z4 = {0, 0, 0, 0};
  v8i z8 = {0, 0, 0, 0, 0, 0, 0, 0};
  __builtin_amdgcn_tensor_load_to_lds(g0, g1, z4, z4, z8, 0);
}

// ---------------------------------------------------------------------------
// K1: mem_out = BETA*mem + x @ W^T.
// Workgroup = 4 waves, 64x64 output tile; per K-step (32) the x-tile (64x32)
// and W-tile (64x32) are DMA'd into LDS by the TDM (double-buffered, issued
// by wave 0, synchronized with s_wait_tensorcnt + barriers); all waves then
// feed bf16 WMMA from LDS.
// ---------------------------------------------------------------------------
__global__ __launch_bounds__(128) void k1_gemm_memout(
    const float* __restrict__ x, const float* __restrict__ W,
    const float* __restrict__ mem, float* __restrict__ memout) {
  __shared__ float smem[4 * TILE_FLTS];   // xbuf0,xbuf1,wbuf0,wbuf1 (36 KiB)

  const int tid = threadIdx.x;
  const int wv  = tid >> 5;          // wave 0..3
  const int ln  = tid & 31;
  const int h   = ln >> 4;
  const int r16 = ln & 15;
  const int wy  = wv >> 1;           // M half
  const int wx  = wv & 1;            // N half
  const int n0  = blockIdx.x * 64;   // N block base
  const int m0  = blockIdx.y * 64;   // B block base
  const int gm  = m0 + wy * 32;      // this wave's 32x32 tile
  const int gn  = n0 + wx * 32;

  const unsigned lds_base = (unsigned)(unsigned long long)(uintptr_t)(void*)smem;
  const float* xblk = x + (size_t)m0 * LDIM;
  const float* wblk = W + (size_t)n0 * LDIM;

  v8f acc00 = {}, acc01 = {}, acc10 = {}, acc11 = {};

  const int KSTEPS = LDIM / TK;      // 128
  if (tid < 32) {                    // wave 0 drives the TDM
    tdm_load_2d(xblk, lds_base, TROWS, TK, LDIM);
    tdm_load_2d(wblk, lds_base + 2u * TILE_FLTS * 4u, TROWS, TK, LDIM);
  }

  for (int i = 0; i < KSTEPS; ++i) {
    const int cur = i & 1;
    if (tid < 32) {
      if (i + 1 < KSTEPS) {          // prefetch next buffers, keep 2 in flight
        const int nxt = cur ^ 1;
        tdm_load_2d(xblk + (size_t)(i + 1) * TK,
                    lds_base + (unsigned)nxt * TILE_FLTS * 4u, TROWS, TK, LDIM);
        tdm_load_2d(wblk + (size_t)(i + 1) * TK,
                    lds_base + (unsigned)(2 + nxt) * TILE_FLTS * 4u, TROWS, TK, LDIM);
        __builtin_amdgcn_s_wait_tensorcnt(2);
      } else {
        __builtin_amdgcn_s_wait_tensorcnt(0);
      }
    }
    __syncthreads();                 // current buffers visible to all waves

    const float* xb = smem + cur * TILE_FLTS;
    const float* wb = smem + (2 + cur) * TILE_FLTS;
    v16bf a0 = load_row_f32_as_bf16(xb + (wy * 32 + r16) * TSTRIDE, h);
    v16bf a1 = load_row_f32_as_bf16(xb + (wy * 32 + 16 + r16) * TSTRIDE, h);
    v16bf b0 = load_row_f32_as_bf16(wb + (wx * 32 + r16) * TSTRIDE, h);
    v16bf b1 = load_row_f32_as_bf16(wb + (wx * 32 + 16 + r16) * TSTRIDE, h);
    acc00 = wmma_bf16(a0, b0, acc00);
    acc01 = wmma_bf16(a0, b1, acc01);
    acc10 = wmma_bf16(a1, b0, acc10);
    acc11 = wmma_bf16(a1, b1, acc11);

    __syncthreads();                 // done reading before buffers are refilled
  }

  // C/D layout: element i at lane ln -> (M = i + 8*(ln>>4), N = ln&15)
#pragma unroll
  for (int i = 0; i < 8; ++i) {
    int rr0 = gm + i + 8 * h;
    int rr1 = rr0 + 16;
    int c0 = gn + r16, c1 = c0 + 16;
    memout[(size_t)rr0 * NDIM + c0] = BETA * mem[(size_t)rr0 * NDIM + c0] + acc00[i];
    memout[(size_t)rr0 * NDIM + c1] = BETA * mem[(size_t)rr0 * NDIM + c1] + acc01[i];
    memout[(size_t)rr1 * NDIM + c0] = BETA * mem[(size_t)rr1 * NDIM + c0] + acc10[i];
    memout[(size_t)rr1 * NDIM + c1] = BETA * mem[(size_t)rr1 * NDIM + c1] + acc11[i];
  }
}

// ---------------------------------------------------------------------------
// K2: per-row reductions -> dL[b]
// ---------------------------------------------------------------------------
__global__ __launch_bounds__(256) void k2_rowstats(
    const float* __restrict__ memout, const float* __restrict__ out_trace,
    const int* __restrict__ rand_idx, const float* __restrict__ input_sum_p,
    float* __restrict__ dL) {
  __shared__ float sps[256], sns[256];
  const int b = blockIdx.x;
  const int t = threadIdx.x;
  const int j = rand_idx[b];
  float sp = 0.f, sn = 0.f;
  for (int k = t; k < NDIM; k += 256) {
    float mo = memout[(size_t)b * NDIM + k];
    float spike = (mo > 1.0f) ? 1.0f : 0.0f;
    sp += spike * out_trace[(size_t)b * NDIM + k];
    sn += spike * out_trace[(size_t)j * NDIM + k];
  }
  sps[t] = sp; sns[t] = sn;
  __syncthreads();
  for (int s = 128; s > 0; s >>= 1) {
    if (t < s) { sps[t] += sps[t + s]; sns[t] += sns[t + s]; }
    __syncthreads();
  }
  if (t == 0) {
    float is = *input_sum_p;
    float loss = GAMMA * is - sps[0] / (float)NDIM + sns[0] / (float)NDIM;
    dL[b] = (loss > 0.0f && is > INPUT_THR) ? 1.0f : 0.0f;
  }
}

// ---------------------------------------------------------------------------
// K3: pre-transposed bf16 operands for GEMM #2.
// ---------------------------------------------------------------------------
__global__ __launch_bounds__(256) void k3_prep(
    const float* __restrict__ memout, const float* __restrict__ out_trace,
    const int* __restrict__ rand_idx, const float* __restrict__ dL,
    const float* __restrict__ in_trace, const float* __restrict__ x,
    bf16_t* __restrict__ coefT, bf16_t* __restrict__ trT) {
  const int idx = blockIdx.x * 256 + threadIdx.x;
  const int b = idx >> 12;
  const int c = idx & 4095;
  const int j = rand_idx[b];

  float mo = memout[(size_t)b * NDIM + c];
  float d = PI_F * (mo - 1.0f);
  float surr = 1.0f / (1.0f + d * d);
  float e = out_trace[(size_t)b * NDIM + c] - out_trace[(size_t)j * NDIM + c];
  coefT[(size_t)c * BSZ + b] = (__bf16)(-dL[b] * e * surr);

  float t = BETA * in_trace[idx] + x[idx];
  trT[(size_t)c * BSZ + b] = (__bf16)t;
}

// ---------------------------------------------------------------------------
// K4: dW over K=BSZ, fused W_new = W - (LR/(B*N)) * dW.
// ---------------------------------------------------------------------------
__global__ __launch_bounds__(32) void k4_gemm_update(
    const bf16_t* __restrict__ coefT, const bf16_t* __restrict__ trT,
    const float* __restrict__ W, float* __restrict__ Wout) {
  const int ln = threadIdx.x;
  const int h  = ln >> 4;
  const int r16 = ln & 15;
  const int n0 = blockIdx.x * 32;  // L columns
  const int m0 = blockIdx.y * 32;  // N rows

  v8f acc00 = {}, acc01 = {}, acc10 = {}, acc11 = {};
  const bf16_t* ar0 = coefT + (size_t)(m0 + r16) * BSZ;
  const bf16_t* ar1 = coefT + (size_t)(m0 + 16 + r16) * BSZ;
  const bf16_t* br0 = trT + (size_t)(n0 + r16) * BSZ;
  const bf16_t* br1 = trT + (size_t)(n0 + 16 + r16) * BSZ;

#pragma unroll
  for (int k0 = 0; k0 < BSZ; k0 += 32) {
    v16bf a0 = load_row_bf16(ar0 + k0, h);
    v16bf a1 = load_row_bf16(ar1 + k0, h);
    v16bf b0 = load_row_bf16(br0 + k0, h);
    v16bf b1 = load_row_bf16(br1 + k0, h);
    acc00 = wmma_bf16(a0, b0, acc00);
    acc01 = wmma_bf16(a0, b1, acc01);
    acc10 = wmma_bf16(a1, b0, acc10);
    acc11 = wmma_bf16(a1, b1, acc11);
  }

  const float scale = LR / ((float)BSZ * (float)NDIM);
#pragma unroll
  for (int i = 0; i < 8; ++i) {
    int rr0 = m0 + i + 8 * h;
    int rr1 = rr0 + 16;
    int c0 = n0 + r16, c1 = c0 + 16;
    Wout[(size_t)rr0 * LDIM + c0] = W[(size_t)rr0 * LDIM + c0] - scale * acc00[i];
    Wout[(size_t)rr0 * LDIM + c1] = W[(size_t)rr0 * LDIM + c1] - scale * acc01[i];
    Wout[(size_t)rr1 * LDIM + c0] = W[(size_t)rr1 * LDIM + c0] - scale * acc10[i];
    Wout[(size_t)rr1 * LDIM + c1] = W[(size_t)rr1 * LDIM + c1] - scale * acc11[i];
  }
}

// ---------------------------------------------------------------------------
extern "C" void kernel_launch(void* const* d_in, const int* in_sizes, int n_in,
                              void* d_out, int out_size, void* d_ws, size_t ws_size,
                              hipStream_t stream) {
  const float* x         = (const float*)d_in[0];
  const float* in_trace  = (const float*)d_in[1];
  const float* out_trace = (const float*)d_in[2];
  const float* mem       = (const float*)d_in[3];
  const float* W         = (const float*)d_in[4];
  const float* input_sum = (const float*)d_in[5];
  const int*   rand_idx  = (const int*)d_in[6];
  float* Wout = (float*)d_out;

  char* ws = (char*)d_ws;
  float*  memout = (float*)ws;                                        // 4 MiB
  float*  dL     = (float*)(ws + (size_t)BSZ * NDIM * sizeof(float));
  bf16_t* coefT  = (bf16_t*)(ws + (size_t)BSZ * NDIM * sizeof(float) + 4096);
  bf16_t* trT    = coefT + (size_t)NDIM * BSZ;

  k1_gemm_memout<<<dim3(NDIM / 64, BSZ / 64), 128, 0, stream>>>(x, W, mem, memout);
  k2_rowstats<<<BSZ, 256, 0, stream>>>(memout, out_trace, rand_idx, input_sum, dL);
  k3_prep<<<(BSZ * NDIM) / 256, 256, 0, stream>>>(memout, out_trace, rand_idx, dL,
                                                  in_trace, x, coefT, trT);
  k4_gemm_update<<<dim3(LDIM / 32, NDIM / 32), 32, 0, stream>>>(coefT, trT, W, Wout);
}
